// AttentionBlock_71262097375323
// MI455X (gfx1250) — matrix operand order repaired and checked by reading, not compiled, not run on hardware
//
#include <hip/hip_runtime.h>

typedef _Float16 half_t;
typedef __attribute__((ext_vector_type(16))) _Float16 v16h;
typedef __attribute__((ext_vector_type(8)))  _Float16 v8h;
typedef __attribute__((ext_vector_type(8)))  float    v8f;

constexpr int HD    = 16;
constexpr int Bb    = 4;
constexpr int Tt    = 250;
constexpr int Ff    = 257;
constexpr int TF    = Tt * Ff;       // 64250
constexpr int S_TOT = Bb * TF;       // 257000 sites
constexpr int SITES = 16;            // sites per block (one WMMA M-tile)
constexpr float SCALE  = 0.25f;      // HD^-0.5
constexpr float EPS_LN = 1e-5f;

// ---- LDS layout (~99 KB; 320 KB/WGP available -> 3 blocks/WGP) ----
// All WMMA operands live in per-lane fragment order: [kTile][lane][16 halves],
// so a fragment load is two aligned ds_load_b128.
struct __align__(16) SMem {
  half_t Wf[4][4][2][512];   // [mat][nTile][kTile][lane*16+e]  (32 KB)
  half_t Afq[2][512];        // LN(q) A fragments                (2 KB)
  half_t Afkv[4][2][512];    // LN(kv) A fragments per channel   (8 KB)
  half_t Afo[2][512];        // attention-out A fragments        (2 KB)
  float  xs[64][65];         // staged x: row = c*16+site        (16.6 KB)
  float  qbuf[16][65];       // q features; reused as out buffer
  float  kbuf[4][16][65];
  float  vbuf[4][16][65];
  float  vecs[8][64];        // gq,bq,gkv,bkv,bWq,bWk,bWv,bWo
};

// Row-major (row, dcol) -> offset inside a [2][512] fragment buffer.
// ISA 7.12.2: lane = g*16+row holds K = kt*32 + g*8 + e + (e>=8 ? 8 : 0).
__device__ inline int frag_off(int row, int dcol) {
  const int kt = dcol >> 5;
  const int dd = dcol & 31;
  const int g  = (dd >> 3) & 1;
  const int e  = (dd & 7) + ((dd >> 4) << 3);
  return kt * 512 + (((g << 4) + row) << 4) + e;
}

// Fragment load: two 16B LDS loads per lane (32B-aligned).
__device__ inline v16h load_frag(const half_t* base, int lane) {
  const v8h* p = (const v8h*)(base + lane * 16);
  const v8h lo = p[0];
  const v8h hi = p[1];
  return __builtin_shufflevector(lo, hi, 0, 1, 2, 3, 4, 5, 6, 7,
                                 8, 9, 10, 11, 12, 13, 14, 15);
}

// One 16x16 output tile, K=64 -> two v_wmma_f32_16x16x32_f16.
__device__ inline v8f wmma_tile64(const half_t* Af, const half_t* Bf, int lane) {
  v8f acc = {};
  v16h a0 = load_frag(Af, lane);
  v16h b0 = load_frag(Bf, lane);
  acc = __builtin_amdgcn_wmma_f32_16x16x32_f16(false, a0, false, b0,
                                               (short)0, acc, false, false);
  v16h a1 = load_frag(Af + 512, lane);
  v16h b1 = load_frag(Bf + 512, lane);
  acc = __builtin_amdgcn_wmma_f32_16x16x32_f16(false, a1, false, b1,
                                               (short)0, acc, false, false);
  return acc;
}

// D f32 16x16: element r -> row M = (lane>>4)*8 + r, col N = lane&15.
__device__ inline void store_D_bias(float* Dst, int ldd, int col0, v8f acc,
                                    const float* bias, int lane) {
  const int n  = lane & 15;
  const int mb = (lane >> 4) << 3;
  const float bv = bias[col0 + n];
#pragma unroll
  for (int r = 0; r < 8; ++r) Dst[(mb + r) * ldd + col0 + n] = acc[r] + bv;
}

__global__ __launch_bounds__(128)
void AttentionBlock_kernel(const float* __restrict__ x,
                           const float* __restrict__ gq,  const float* __restrict__ bq,
                           const float* __restrict__ gkv, const float* __restrict__ bkv,
                           const float* __restrict__ Wq,  const float* __restrict__ bWq,
                           const float* __restrict__ Wk,  const float* __restrict__ bWk,
                           const float* __restrict__ Wv,  const float* __restrict__ bWv,
                           const float* __restrict__ Wo,  const float* __restrict__ bWo,
                           float* __restrict__ out) {
  __shared__ SMem sm;
  const int tid  = threadIdx.x;
  const int lane = tid & 31;
  const int w    = tid >> 5;                   // wave id 0..3
  const int s0   = blockIdx.x * SITES;

  // --- stage weights directly in B-fragment order (wave w -> matrix w) ---
  // B[k][n] = W[n][k]; fragment lane = g*16 + (n&15) holds K = kt*32 + g*8 + e + (e>=8 ? 8 : 0).
  const float* Wsrc = (w == 0) ? Wq : (w == 1) ? Wk : (w == 2) ? Wv : Wo;
#pragma unroll 4
  for (int i = 0; i < 128; ++i) {
    const int j = lane + 32 * i;               // coalesced over n*64+k
    const int k = j & 63, n = j >> 6;
    const int kt = k >> 5, g = (k >> 3) & 1, e = (k & 7) + (((k >> 4) & 1) << 3);
    sm.Wf[w][n >> 4][kt][(((g << 4) | (n & 15)) << 4) + e] = (half_t)Wsrc[j];
  }
  // --- stage LN gains/biases and projection biases ---
  if (tid < 64) {
    sm.vecs[0][tid] = gq[tid];  sm.vecs[1][tid] = bq[tid];
    sm.vecs[2][tid] = gkv[tid]; sm.vecs[3][tid] = bkv[tid];
    sm.vecs[4][tid] = bWq[tid]; sm.vecs[5][tid] = bWk[tid];
    sm.vecs[6][tid] = bWv[tid]; sm.vecs[7][tid] = bWo[tid];
  }
  // --- stage x: lanes sweep consecutive f (contiguous dim) -> coalesced ---
#pragma unroll 4
  for (int i = 0; i < 32; ++i) {
    const int j    = tid + 128 * i;            // 0..4095
    const int site = j & 15;
    const int cd   = j >> 4;
    const int d    = cd & 63, c = cd >> 6;
    const int s    = s0 + site;
    float v = 0.f;
    if (s < S_TOT) {
      const int b = s / TF;
      v = x[(size_t)s + (size_t)(255 * b + 64 * c + d) * TF];
    }
    sm.xs[c * 16 + site][d] = v;
  }
  __syncthreads();

  // --- layernorm: one row per lane (64 rows), emit f16 A fragments ---
  if (tid < 64) {
    const int c  = tid >> 4;
    const int rr = tid & 15;
    const float* xr = sm.xs[tid];
    float mu = 0.f;
#pragma unroll
    for (int d = 0; d < 64; ++d) mu += xr[d];
    mu *= (1.f / 64.f);
    float var = 0.f;
#pragma unroll
    for (int d = 0; d < 64; ++d) { const float t = xr[d] - mu; var += t * t; }
    var *= (1.f / 64.f);
    const float rs = rsqrtf(var + EPS_LN);
    half_t* akv = &sm.Afkv[c][0][0];
    half_t* aq  = &sm.Afq[0][0];
    const bool isC0 = (tid < 16);
#pragma unroll
    for (int d = 0; d < 64; ++d) {
      const float nrm = (xr[d] - mu) * rs;
      const int  off  = frag_off(rr, d);
      akv[off] = (half_t)(nrm * sm.vecs[2][d] + sm.vecs[3][d]);
      if (isC0) aq[off] = (half_t)(nrm * sm.vecs[0][d] + sm.vecs[1][d]);
    }
  }
  __syncthreads();

  // --- Q/K/V projections via WMMA ---
  // wave w: Q n-tile w; K and V for channel c=w (all 4 n-tiles)
  {
    v8f accq = wmma_tile64(&sm.Afq[0][0], &sm.Wf[0][w][0][0], lane);
    store_D_bias(&sm.qbuf[0][0], 65, w * 16, accq, sm.vecs[4], lane);
#pragma unroll
    for (int nt = 0; nt < 4; ++nt) {
      v8f acck = wmma_tile64(&sm.Afkv[w][0][0], &sm.Wf[1][nt][0][0], lane);
      store_D_bias(&sm.kbuf[w][0][0], 65, nt * 16, acck, sm.vecs[5], lane);
      v8f accv = wmma_tile64(&sm.Afkv[w][0][0], &sm.Wf[2][nt][0][0], lane);
      store_D_bias(&sm.vbuf[w][0][0], 65, nt * 16, accv, sm.vecs[6], lane);
    }
  }
  __syncthreads();

  // --- attention: one (site, head) pair per lane (4 keys, HD=16) ---
  if (tid < 64) {
    const int site = tid & 15, h = tid >> 4;
    const int f0 = h * HD;
    float qv[HD];
#pragma unroll
    for (int d = 0; d < HD; ++d) qv[d] = sm.qbuf[site][f0 + d];
    float sc[4];
#pragma unroll
    for (int c = 0; c < 4; ++c) {
      float dot = 0.f;
#pragma unroll
      for (int d = 0; d < HD; ++d) dot += qv[d] * sm.kbuf[c][site][f0 + d];
      sc[c] = dot * SCALE;
    }
    float m = sc[0];
#pragma unroll
    for (int c = 1; c < 4; ++c) m = fmaxf(m, sc[c]);
    float e[4], sum = 0.f;
#pragma unroll
    for (int c = 0; c < 4; ++c) { e[c] = __expf(sc[c] - m); sum += e[c]; }
    const float inv = 1.f / sum;
    half_t* aof = &sm.Afo[0][0];
#pragma unroll
    for (int d = 0; d < HD; ++d) {
      float o = 0.f;
#pragma unroll
      for (int c = 0; c < 4; ++c) o += e[c] * sm.vbuf[c][site][f0 + d];
      aof[frag_off(site, f0 + d)] = (half_t)(o * inv);
    }
  }
  __syncthreads();

  // --- output projection via WMMA (reuse qbuf as out buffer) ---
  {
    v8f acco = wmma_tile64(&sm.Afo[0][0], &sm.Wf[3][w][0][0], lane);
    store_D_bias(&sm.qbuf[0][0], 65, w * 16, acco, sm.vecs[7], lane);
  }
  __syncthreads();

  // --- residual + coalesced global store (lanes sweep consecutive f) ---
#pragma unroll 4
  for (int i = 0; i < 8; ++i) {
    const int j    = tid + 128 * i;            // 0..1023
    const int site = j & 15;
    const int d    = j >> 4;
    const int s    = s0 + site;
    if (s < S_TOT) {
      const int b = s / TF;
      out[(size_t)s + (size_t)(63 * b + d) * TF] = sm.qbuf[site][d] + sm.xs[site][d];
    }
  }
}

extern "C" void kernel_launch(void* const* d_in, const int* in_sizes, int n_in,
                              void* d_out, int out_size, void* d_ws, size_t ws_size,
                              hipStream_t stream) {
  (void)in_sizes; (void)n_in; (void)out_size; (void)d_ws; (void)ws_size;
  const float* x   = (const float*)d_in[0];
  const float* gq  = (const float*)d_in[1];
  const float* bq  = (const float*)d_in[2];
  const float* gkv = (const float*)d_in[3];
  const float* bkv = (const float*)d_in[4];
  const float* Wq  = (const float*)d_in[5];
  const float* bWq = (const float*)d_in[6];
  const float* Wk  = (const float*)d_in[7];
  const float* bWk = (const float*)d_in[8];
  const float* Wv  = (const float*)d_in[9];
  const float* bWv = (const float*)d_in[10];
  const float* Wo  = (const float*)d_in[11];
  const float* bWo = (const float*)d_in[12];
  float* out = (float*)d_out;

  const int nblk = (S_TOT + SITES - 1) / SITES;   // 16063 blocks
  AttentionBlock_kernel<<<nblk, 128, 0, stream>>>(
      x, gq, bq, gkv, bkv, Wq, bWq, Wk, bWk, Wv, bWv, Wo, bWo, out);
}
